// WindowedSelfAttention_49022756716936
// MI455X (gfx1250) — matrix operand order repaired and checked by reading, not compiled, not run on hardware
//
#include <hip/hip_runtime.h>

// Windowed self-attention (Swin), fully fused: one workgroup per 49-token window.
// f16 WMMA (v_wmma_f32_16x16x32_f16) with f32 accumulate; softmax in f32.

#define NTOK 49
#define NPAD 64
#define DIMC 128
#define HEADS 4
#define HD 32
#define NW 64

typedef __attribute__((ext_vector_type(16))) _Float16 v16h;
typedef __attribute__((ext_vector_type(8)))  float    v8f;
typedef __attribute__((ext_vector_type(8)))  _Float16 h8;

// element strides (chosen so every vector LDS access is 16B-aligned)
#define XH_RS   136   // x tile   f16 [64][136]
#define QK_RS   40    // q,k      f16 [4][64][40]
#define VT_RS   72    // v^T      f16 [4][32][72]
#define AS_RS   65    // scores   f32 [4][64][65]
#define AH_RS   72    // probs    f16 [4][64][72]
#define Y_RS    136   // y tile   f16 [64][136]

// shared-memory layout (bytes); regions re-used across phases
#define OFF_A     0                          // union: xh (ph0-1) / scores (ph2-3) / y (ph4-5)
#define SZ_A      (HEADS*NPAD*AS_RS*4)       // 66560
#define OFF_QA    (OFF_A + SZ_A)             // union: q (ph1-2) / probs (ph3-4)
#define SZ_QA     (HEADS*NPAD*AH_RS*2)       // 36864
#define OFF_K     (OFF_QA + SZ_QA)           // k
#define SZ_K      (HEADS*NPAD*QK_RS*2)       // 20480
#define OFF_VT    (OFF_K + SZ_K)             // v transposed
#define SZ_VT     (HEADS*HD*VT_RS*2)         // 18432
#define OFF_BT    (OFF_VT + SZ_VT)           // bias table
#define SZ_BT     (225*HEADS*4)              // 3600
#define OFF_MSK   (OFF_BT + SZ_BT)           // 145936: window mask (f32 [49*49])
#define SZ_MSK    ((NTOK*NTOK*4 + 15) & ~15) // 9616
#define SMEM_BYTES (OFF_MSK + SZ_MSK)        // 155552 (< 320KB/WGP; 2 WGs fit)

__device__ __forceinline__ v8f wmma16(v16h a, v16h b, v8f c) {
  return __builtin_amdgcn_wmma_f32_16x16x32_f16(false, a, false, b, (short)0, c,
                                                false, false);
}

// A-fragment (16x32 f16, MxK): lane<16 holds row M=lane, K = k0+0..15;
// lane>=16 holds row M=lane-16, K = k0+16..31. Storage: S[m][k], k contig.
__device__ __forceinline__ v16h load_a(const _Float16* S, int m0, int k0, int rs,
                                       int lane) {
  int sel = lane >> 4, mr = lane & 15;
  const _Float16* p = S + (m0 + mr) * rs + k0 + sel * 16;
  h8 lo = *(const h8*)p;
  h8 hi = *(const h8*)(p + 8);
  v16h r;
#pragma unroll
  for (int i = 0; i < 8; ++i) { r[i] = lo[i]; r[i + 8] = hi[i]; }
  return r;
}

// B-fragment (32x16 f16, KxN) from N-major storage S[n][k] (k contiguous):
// lane<16: col N=n0+lane, K=k0+0..15 ; lane>=16: col N=n0+lane-16, K=k0+16..31.
__device__ __forceinline__ v16h load_b(const _Float16* S, int n0, int k0, int rs,
                                       int lane) {
  int sel = lane >> 4, nr = lane & 15;
  const _Float16* p = S + (n0 + nr) * rs + k0 + sel * 16;
  h8 lo = *(const h8*)p;
  h8 hi = *(const h8*)(p + 8);
  v16h r;
#pragma unroll
  for (int i = 0; i < 8; ++i) { r[i] = lo[i]; r[i + 8] = hi[i]; }
  return r;
}

// B-fragment from a row-major f32 weight matrix W[n][k] in global memory.
__device__ __forceinline__ v16h load_b_gf32(const float* W, int n0, int k0, int rs,
                                            int lane) {
  int sel = lane >> 4, nr = lane & 15;
  const float* p = W + (n0 + nr) * rs + k0 + sel * 16;
  v16h r;
#pragma unroll
  for (int i = 0; i < 16; ++i) r[i] = (_Float16)p[i];
  return r;
}

__global__ void __launch_bounds__(256)
swin_window_attn_kernel(const float* __restrict__ x, const float* __restrict__ mask,
                        const float* __restrict__ qkv_w, const float* __restrict__ qkv_b,
                        const float* __restrict__ proj_w, const float* __restrict__ proj_b,
                        const float* __restrict__ bias_table,
                        const int* __restrict__ rel_index, float* __restrict__ out) {
  __shared__ __align__(16) unsigned char smem[SMEM_BYTES];
  _Float16* xh    = (_Float16*)(smem + OFF_A);
  float*    scs   = (float*)(smem + OFF_A);
  _Float16* yld   = (_Float16*)(smem + OFF_A);
  _Float16* qld   = (_Float16*)(smem + OFF_QA);
  _Float16* ah    = (_Float16*)(smem + OFF_QA);
  _Float16* kld   = (_Float16*)(smem + OFF_K);
  _Float16* vtld  = (_Float16*)(smem + OFF_VT);
  float*    btab  = (float*)(smem + OFF_BT);
  float*    mlds  = (float*)(smem + OFF_MSK);

  const int tid  = threadIdx.x;
  const int lane = tid & 31;
  const int wave = tid >> 5;
  const int sel  = lane >> 4;
  const int nr   = lane & 15;
  const int blk  = blockIdx.x;
  const int wm   = blk & (NW - 1);           // mask window index (B_ / nW groups)
  const float scale = 0.17677669529663689f;  // HD^-0.5

  // ---- speculative prefetch of the (L2-resident) weight matrices -----------
  // one byte per 128B cacheline -> global_prefetch_b8; no LOADcnt cost.
  for (int off = tid * 128; off < 3 * DIMC * DIMC * 4; off += 256 * 128)
    __builtin_prefetch((const char*)qkv_w + off, 0, 0);
  for (int off = tid * 128; off < DIMC * DIMC * 4; off += 256 * 128)
    __builtin_prefetch((const char*)proj_w + off, 0, 0);

  // ---------------- phase 0: stage x (f16), bias table, mask into LDS -------
  const float* xg = x + (size_t)blk * NTOK * DIMC;
  for (int i = tid; i < NTOK * DIMC; i += 256)
    xh[(i / DIMC) * XH_RS + (i % DIMC)] = (_Float16)xg[i];
  for (int i = tid; i < (NPAD - NTOK) * DIMC; i += 256)
    xh[(NTOK + i / DIMC) * XH_RS + (i % DIMC)] = (_Float16)0.f;
  for (int i = tid; i < 225 * HEADS; i += 256) btab[i] = bias_table[i];
  {
    const float* mg = mask + (size_t)wm * NTOK * NTOK;
    for (int i = tid; i < NTOK * NTOK; i += 256) mlds[i] = mg[i];
  }
  __syncthreads();

  // ---------------- phase 1: qkv = x @ qkv_w^T + qkv_b ----------------------
  // M=64(pad) x N=384 x K=128 ; 24 N-tiles striped over 8 waves.
  for (int nt = wave; nt < 24; nt += 8) {
    const int n0 = nt * 16;
    v8f acc[4] = {};
#pragma unroll
    for (int ks = 0; ks < 4; ++ks) {
      const int k0 = ks * 32;
      v16h b = load_b_gf32(qkv_w, n0, k0, DIMC, lane);
#pragma unroll
      for (int mt = 0; mt < 4; ++mt)
        acc[mt] = wmma16(load_a(xh, mt * 16, k0, XH_RS, lane), b, acc[mt]);
    }
    const float bb   = qkv_b[n0 + nr];
    const int which  = nt >> 3;          // 0=q 1=k 2=v
    const int h      = (nt & 7) >> 1;
    const int d0     = (nt & 1) * 16;
#pragma unroll
    for (int mt = 0; mt < 4; ++mt) {
      if (which == 2) {                  // v stored transposed: vT[h][d][t]
        h8 pk;
#pragma unroll
        for (int r = 0; r < 8; ++r) pk[r] = (_Float16)(acc[mt][r] + bb);
        const int tb = mt * 16 + sel * 8;
        *(h8*)(vtld + (h * HD + d0 + nr) * VT_RS + tb) = pk;
      } else {                           // q (pre-scaled) / k stored [h][t][d]
        _Float16* dst = (which == 0 ? qld : kld) + h * NPAD * QK_RS + d0 + nr;
        const float s = (which == 0) ? scale : 1.f;
#pragma unroll
        for (int r = 0; r < 8; ++r) {
          const int t = mt * 16 + sel * 8 + r;
          dst[t * QK_RS] = (_Float16)((acc[mt][r] + bb) * s);
        }
      }
    }
  }
  __syncthreads();

  // ---------------- phase 2: S = (q*scale) @ k^T + bias + mask --------------
  {
    const int h  = wave >> 1;
    const int mh = wave & 1;
    const _Float16* qh = qld + h * NPAD * QK_RS;
    const _Float16* kh = kld + h * NPAD * QK_RS;
    float* sh = scs + h * NPAD * AS_RS;
#pragma unroll
    for (int mi = 0; mi < 2; ++mi) {
      const int mt = mh * 2 + mi;
      v16h a = load_a(qh, mt * 16, 0, QK_RS, lane);
#pragma unroll
      for (int ntj = 0; ntj < 4; ++ntj) {
        v8f c = {};
        c = wmma16(a, load_b(kh, ntj * 16, 0, QK_RS, lane), c);
        const int j = ntj * 16 + nr;
#pragma unroll
        for (int r = 0; r < 8; ++r) {
          const int i = mt * 16 + sel * 8 + r;
          float val = c[r];
          if (i < NTOK && j < NTOK)
            val += btab[rel_index[i * NTOK + j] * HEADS + h] + mlds[i * NTOK + j];
          else
            val = -1e30f;
          sh[i * AS_RS + j] = val;
        }
      }
    }
  }
  __syncthreads();

  // ---------------- phase 3: row softmax (f32), write f16 probs -------------
  {
    const int h = tid >> 6;
    const int i = tid & 63;
    _Float16* arow = ah + (h * NPAD + i) * AH_RS;
    if (i < NTOK) {
      const float* srow = scs + (h * NPAD + i) * AS_RS;
      float m = -1e30f;
      for (int j = 0; j < NTOK; ++j) m = fmaxf(m, srow[j]);
      float s = 0.f;
      for (int j = 0; j < NTOK; ++j) s += __expf(srow[j] - m);
      const float inv = 1.f / s;
      for (int j = 0; j < NTOK; ++j) arow[j] = (_Float16)(__expf(srow[j] - m) * inv);
      for (int j = NTOK; j < NPAD; ++j) arow[j] = (_Float16)0.f;  // K padding
    } else {
      for (int j = 0; j < NPAD; ++j) arow[j] = (_Float16)0.f;
    }
  }
  __syncthreads();

  // ---------------- phase 4: y = probs @ v ----------------------------------
  {
    const int h  = wave >> 1;
    const int mh = wave & 1;
    const _Float16* ahh = ah + h * NPAD * AH_RS;
    const _Float16* vth = vtld + h * HD * VT_RS;
#pragma unroll
    for (int mi = 0; mi < 2; ++mi) {
      const int mt = mh * 2 + mi;
#pragma unroll
      for (int ntd = 0; ntd < 2; ++ntd) {
        v8f c = {};
#pragma unroll
        for (int ks = 0; ks < 2; ++ks)
          c = wmma16(load_a(ahh, mt * 16, ks * 32, AH_RS, lane),
                     load_b(vth, ntd * 16, ks * 32, VT_RS, lane), c);
        const int cidx = h * HD + ntd * 16 + nr;
#pragma unroll
        for (int r = 0; r < 8; ++r) {
          const int t = mt * 16 + sel * 8 + r;
          yld[t * Y_RS + cidx] = (_Float16)c[r];
        }
      }
    }
  }
  __syncthreads();

  // ---------------- phase 5: out = y @ proj_w^T + proj_b --------------------
  {
    const int n0 = wave * 16;  // 8 N-tiles of 16 over 8 waves
    v8f acc[4] = {};
#pragma unroll
    for (int ks = 0; ks < 4; ++ks) {
      const int k0 = ks * 32;
      v16h b = load_b_gf32(proj_w, n0, k0, DIMC, lane);
#pragma unroll
      for (int mt = 0; mt < 4; ++mt)
        acc[mt] = wmma16(load_a(yld, mt * 16, k0, Y_RS, lane), b, acc[mt]);
    }
    const float bb = proj_b[n0 + nr];
    float* og = out + (size_t)blk * NTOK * DIMC;
#pragma unroll
    for (int mt = 0; mt < 4; ++mt) {
#pragma unroll
      for (int r = 0; r < 8; ++r) {
        const int t = mt * 16 + sel * 8 + r;
        if (t < NTOK) og[t * DIMC + n0 + nr] = acc[mt][r] + bb;
      }
    }
  }
}

extern "C" void kernel_launch(void* const* d_in, const int* in_sizes, int n_in,
                              void* d_out, int out_size, void* d_ws, size_t ws_size,
                              hipStream_t stream) {
  const float* x          = (const float*)d_in[0];
  const float* mask       = (const float*)d_in[1];
  const float* qkv_w      = (const float*)d_in[2];
  const float* qkv_b      = (const float*)d_in[3];
  const float* proj_w     = (const float*)d_in[4];
  const float* proj_b     = (const float*)d_in[5];
  const float* bias_table = (const float*)d_in[6];
  const int*   rel_index  = (const int*)d_in[7];
  float* out = (float*)d_out;

  const int B = in_sizes[0] / (NTOK * DIMC);  // 4096 windows
  swin_window_attn_kernel<<<B, 256, 0, stream>>>(x, mask, qkv_w, qkv_b, proj_w,
                                                 proj_b, bias_table, rel_index, out);
}
